// PolygonMatchingNet_26938034880618
// MI455X (gfx1250) — compile-verified, matrix operand order checked by reading, not run on hardware
//
#include <hip/hip_runtime.h>
#include <hip/hip_bf16.h>
#include <stdint.h>
#include <stddef.h>

typedef __attribute__((ext_vector_type(16))) _Float16 v16h;
typedef __attribute__((ext_vector_type(8)))  _Float16 h8;
typedef __attribute__((ext_vector_type(8)))  float    v8f;

// ---------------------------------------------------------------------------
// WMMA GEMM: C[M,Nout] = act( A[M,K] @ WT^T + bias + addsrc )
//  - A:  fp32 row-major (lda), optionally relu'ed on load
//  - WT: f16 [Nout, K] row-major (W transposed), K % 32 == 0, rows 64B aligned
//  - one wave computes a 16x32 tile: one A fragment, two B fragments,
//    2x v_wmma_f32_16x16x32_f16 per K-step
//  - requires M % 16 == 0, Nout % 32 == 0 (true here: M=50000=16*3125)
//  - flags are template params: no runtime branches in the hot path
// ---------------------------------------------------------------------------
template <bool RELU_IN, bool HAS_BIAS, bool HAS_ADD, bool RELU_OUT>
__global__ __launch_bounds__(128)
void k_wmma_gemm(const float* __restrict__ A, int lda,
                 const _Float16* __restrict__ WT,
                 const float* __restrict__ bias,
                 const float* __restrict__ addsrc, int ld_add,
                 float* __restrict__ C, int ldc, int col_off,
                 int M, int Nout, int K)
{
  const int wave = (blockIdx.x * blockDim.x + threadIdx.x) >> 5;   // lane-uniform
  const int lane = threadIdx.x & 31;
  const int tiles_n = Nout >> 5;                 // 32-wide N tiles
  const int tiles_m = M >> 4;
  if (wave >= tiles_m * tiles_n) return;         // whole-wave exit
  const int tile_m = wave / tiles_n;
  const int tile_n = wave - tile_m * tiles_n;

  const int half_hi = lane >> 4;                 // ISA 16-bit A layout: K group +8
  const int lrow    = lane & 15;

  const float4* __restrict__ Ap =
      (const float4*)(A + (size_t)(tile_m * 16 + lrow) * lda);
  const h8* __restrict__ Bp0 =
      (const h8*)(WT + (size_t)(tile_n * 32 + lrow) * (size_t)K);
  const h8* __restrict__ Bp1 =
      (const h8*)(WT + (size_t)(tile_n * 32 + 16 + lrow) * (size_t)K);

  v8f acc0 = {}, acc1 = {};
  for (int kb = 0; kb < K; kb += 32) {
    const int a4 = (kb >> 2) + (half_hi << 1);   // float4 index of run0
    const int b8 = (kb >> 3) + half_hi;          // h8 index of run0
    // prefetch next K-step of A (global_prefetch_b8; speculative, safe)
    __builtin_prefetch(Ap + a4 + 8, 0, 1);

    float4 a0 = Ap[a4];
    float4 a1 = Ap[a4 + 1];
    float4 a2 = Ap[a4 + 4];
    float4 a3 = Ap[a4 + 5];
    if (RELU_IN) {
      a0.x = fmaxf(a0.x, 0.f); a0.y = fmaxf(a0.y, 0.f);
      a0.z = fmaxf(a0.z, 0.f); a0.w = fmaxf(a0.w, 0.f);
      a1.x = fmaxf(a1.x, 0.f); a1.y = fmaxf(a1.y, 0.f);
      a1.z = fmaxf(a1.z, 0.f); a1.w = fmaxf(a1.w, 0.f);
      a2.x = fmaxf(a2.x, 0.f); a2.y = fmaxf(a2.y, 0.f);
      a2.z = fmaxf(a2.z, 0.f); a2.w = fmaxf(a2.w, 0.f);
      a3.x = fmaxf(a3.x, 0.f); a3.y = fmaxf(a3.y, 0.f);
      a3.z = fmaxf(a3.z, 0.f); a3.w = fmaxf(a3.w, 0.f);
    }
    v16h av;
    av[0]  = (_Float16)a0.x; av[1]  = (_Float16)a0.y;
    av[2]  = (_Float16)a0.z; av[3]  = (_Float16)a0.w;
    av[4]  = (_Float16)a1.x; av[5]  = (_Float16)a1.y;
    av[6]  = (_Float16)a1.z; av[7]  = (_Float16)a1.w;
    av[8]  = (_Float16)a2.x; av[9]  = (_Float16)a2.y;
    av[10] = (_Float16)a2.z; av[11] = (_Float16)a2.w;
    av[12] = (_Float16)a3.x; av[13] = (_Float16)a3.y;
    av[14] = (_Float16)a3.z; av[15] = (_Float16)a3.w;

    h8 b00 = Bp0[b8], b01 = Bp0[b8 + 2];
    h8 b10 = Bp1[b8], b11 = Bp1[b8 + 2];
    v16h bv0, bv1;
#pragma unroll
    for (int i = 0; i < 8; ++i) {
      bv0[i] = b00[i]; bv0[8 + i] = b01[i];
      bv1[i] = b10[i]; bv1[8 + i] = b11[i];
    }
    acc0 = __builtin_amdgcn_wmma_f32_16x16x32_f16(false, av, false, bv0,
                                                  (short)0, acc0, false, false);
    acc1 = __builtin_amdgcn_wmma_f32_16x16x32_f16(false, av, false, bv1,
                                                  (short)0, acc1, false, false);
  }

  // D layout: VGPR r, lanes 0-15 -> (M=r, N=lane); lanes 16-31 -> (M=r+8, N=lane-16)
  const int col0 = tile_n * 32 + lrow;
  const int col1 = col0 + 16;
  const float bb0 = HAS_BIAS ? bias[col0] : 0.0f;
  const float bb1 = HAS_BIAS ? bias[col1] : 0.0f;
#pragma unroll
  for (int r = 0; r < 8; ++r) {
    const int row = tile_m * 16 + r + (half_hi << 3);
    float v0 = acc0[r] + bb0;
    float v1 = acc1[r] + bb1;
    if (HAS_ADD) {
      v0 += addsrc[(size_t)row * ld_add + col0];
      v1 += addsrc[(size_t)row * ld_add + col1];
    }
    if (RELU_OUT) { v0 = fmaxf(v0, 0.f); v1 = fmaxf(v1, 0.f); }
    float* crow = C + (size_t)row * ldc + col_off;
    crow[col0] = v0;
    crow[col1] = v1;
  }
}

// ---------------------------------------------------------------------------
// Weight convert + transpose: W[K,Nout] fp32 -> WT[Nout,K] f16
// ---------------------------------------------------------------------------
__global__ void k_conv_wt(const float* __restrict__ W, _Float16* __restrict__ WT,
                          int K, int Nout)
{
  int idx = blockIdx.x * blockDim.x + threadIdx.x;
  if (idx >= K * Nout) return;
  int k = idx / Nout;
  int n = idx - k * Nout;
  WT[(size_t)n * K + k] = (_Float16)W[idx];
}

// node_lin: feat[:, 0:128] = relu(x[N,3] @ W[3,128] + b)
__global__ void k_node_lin(const float* __restrict__ x, const float* __restrict__ W,
                           const float* __restrict__ b, float* __restrict__ C,
                           int ldc, int N_)
{
  int idx = blockIdx.x * blockDim.x + threadIdx.x;
  if (idx >= N_ * 128) return;
  int n = idx >> 7, j = idx & 127;
  float v = b[j];
  v = fmaf(x[n * 3 + 0], W[0 * 128 + j], v);
  v = fmaf(x[n * 3 + 1], W[1 * 128 + j], v);
  v = fmaf(x[n * 3 + 2], W[2 * 128 + j], v);
  C[(size_t)n * ldc + j] = fmaxf(v, 0.f);
}

// ---------------- degree / norm ----------------
__global__ void k_fill1(float* __restrict__ p, int n) {
  int i = blockIdx.x * blockDim.x + threadIdx.x;
  if (i < n) p[i] = 1.0f;                       // self-loop contribution
}
__global__ void k_deg_count(const int* __restrict__ dst, float* __restrict__ deg, int E_) {
  int i = blockIdx.x * blockDim.x + threadIdx.x;
  if (i < E_) atomicAdd(&deg[dst[i]], 1.0f);
}
__global__ void k_rsqrt_inplace(float* __restrict__ p, int n) {
  int i = blockIdx.x * blockDim.x + threadIdx.x;
  if (i < n) p[i] = rsqrtf(p[i]);               // deg >= 1 always
}

// agg[n,j] = bias[j] + xw[n,j] * dinv[n]^2   (self loop term)
__global__ void k_agg_init(const float* __restrict__ xw, const float* __restrict__ dinv,
                           const float* __restrict__ bias, float* __restrict__ agg, int N_)
{
  int idx = blockIdx.x * blockDim.x + threadIdx.x;
  if (idx >= N_ * 128) return;
  int n = idx >> 7, j = idx & 127;
  float di = dinv[n];
  agg[idx] = fmaf(xw[idx], di * di, bias[j]);
}

// one wave per edge: agg[d,:] += xw[s,:] * dinv[s]*dinv[d]   (128 floats)
__global__ __launch_bounds__(256)
void k_gcn_scatter(const int* __restrict__ src, const int* __restrict__ dst, int E_,
                   const float* __restrict__ dinv, const float* __restrict__ xw,
                   float* __restrict__ agg)
{
  int e    = (blockIdx.x * blockDim.x + threadIdx.x) >> 5;
  int lane = threadIdx.x & 31;
  if (e >= E_) return;
  int s = src[e], d = dst[e];
  float w = dinv[s] * dinv[d];
  float4 v = ((const float4*)(xw + (size_t)s * 128))[lane];
  float* a = agg + (size_t)d * 128 + lane * 4;
  atomicAdd(a + 0, v.x * w);
  atomicAdd(a + 1, v.y * w);
  atomicAdd(a + 2, v.z * w);
  atomicAdd(a + 3, v.w * w);
}

// cat[n, 0:wa] = A[n,:], cat[n, wa:wa+wb] = B[n,:]
__global__ void k_concat2(const float* __restrict__ A, int wa,
                          const float* __restrict__ B, int wb,
                          float* __restrict__ Cc, int N_)
{
  int w = wa + wb;
  int idx = blockIdx.x * blockDim.x + threadIdx.x;
  if (idx >= N_ * w) return;
  int n = idx / w, j = idx - n * w;
  Cc[idx] = (j < wa) ? A[(size_t)n * wa + j] : B[(size_t)n * wb + (j - wa)];
}

// ---------------------------------------------------------------------------
// Host orchestration
// ---------------------------------------------------------------------------
static inline void gemm(hipStream_t s, const float* A, int lda, int relu_in,
                        const _Float16* WT, const float* bias,
                        const float* addsrc, int ld_add,
                        float* C, int ldc, int col_off, int relu_out,
                        int M, int Nout, int K)
{
  int tiles  = (M >> 4) * (Nout >> 5);
  int blocks = (tiles + 3) >> 2;                 // 4 waves (128 threads) per block
  if (relu_in) {
    k_wmma_gemm<true, true, false, false><<<blocks, 128, 0, s>>>(
        A, lda, WT, bias, addsrc, ld_add, C, ldc, col_off, M, Nout, K);
  } else if (addsrc) {
    k_wmma_gemm<false, true, true, true><<<blocks, 128, 0, s>>>(
        A, lda, WT, bias, addsrc, ld_add, C, ldc, col_off, M, Nout, K);
  } else if (!bias) {
    k_wmma_gemm<false, false, false, false><<<blocks, 128, 0, s>>>(
        A, lda, WT, bias, addsrc, ld_add, C, ldc, col_off, M, Nout, K);
  } else if (relu_out) {
    k_wmma_gemm<false, true, false, true><<<blocks, 128, 0, s>>>(
        A, lda, WT, bias, addsrc, ld_add, C, ldc, col_off, M, Nout, K);
  } else {
    k_wmma_gemm<false, true, false, false><<<blocks, 128, 0, s>>>(
        A, lda, WT, bias, addsrc, ld_add, C, ldc, col_off, M, Nout, K);
  }
}

extern "C" void kernel_launch(void* const* d_in, const int* in_sizes, int n_in,
                              void* d_out, int out_size, void* d_ws, size_t ws_size,
                              hipStream_t stream)
{
  const int N = in_sizes[0] / 3;      // 50000
  const int E = in_sizes[3] / 2;      // 600000
  const int G = 64;

  // params flattened (pytree: sorted dict keys, (w,b) order), starting at d_in[8]
  const float* P[36];
  for (int i = 0; i < 36; ++i) P[i] = (const float*)d_in[8 + i];
  const float *fc_w[4],  *fc_b[4],  *gcn_w[4], *gcn_b[4], *proj_w[4], *proj_b[4];
  for (int i = 0; i < 4; ++i) { fc_w[i]  = P[0 + 2*i]; fc_b[i]  = P[1 + 2*i]; }
  for (int i = 0; i < 4; ++i) { gcn_w[i] = P[8 + 2*i]; gcn_b[i] = P[9 + 2*i]; }
  const float* gfc1_w = P[16]; const float* gfc1_b = P[17];
  const float* gfc2_w = P[18]; const float* gfc2_b = P[19];
  const float* gprj_w = P[20]; const float* gprj_b = P[21];
  const float* node_w = P[22]; const float* node_b = P[23];
  const float* pfc1_w = P[24]; const float* pfc1_b = P[25];
  const float* pfc2_w = P[26]; const float* pfc2_b = P[27];
  for (int i = 0; i < 4; ++i) { proj_w[i] = P[28 + 2*i]; proj_b[i] = P[29 + 2*i]; }

  // ---- workspace carve-up ----
  size_t off = 0;
  auto carve = [&](size_t bytes) -> char* {
    char* p = (char*)d_ws + off;
    off += (bytes + 255) & ~(size_t)255;
    return p;
  };
  float* b_feat = (float*)carve((size_t)N * 256 * 4);   // h0 = [node_b | patch_b]
  float* b_cat  = (float*)carve((size_t)N * 320 * 4);   // concat / patch-fc1 tmp
  float* b_xw   = (float*)carve((size_t)N * 128 * 4);
  float* b_agg  = (float*)carve((size_t)N * 128 * 4);
  float* b_out  = (float*)carve((size_t)N * 64  * 4);
  float* b_h    = (float*)carve((size_t)N * 64  * 4);
  float* dinv   = (float*)carve((size_t)N * 4);
  float* g_t1   = (float*)carve((size_t)G * 256 * 4);
  float* g_t2   = (float*)carve((size_t)G * 128 * 4);

  // f16 transposed weights
  struct WSpec { const float* w; int K, Nout; _Float16* t; };
  WSpec WS[] = {
    { pfc1_w, 128, 256, nullptr }, { pfc2_w, 256, 128, nullptr },
    { gfc1_w, 128, 256, nullptr }, { gfc2_w, 256, 128, nullptr },
    { gprj_w, 128,  64, nullptr },
    { gcn_w[0], 256, 128, nullptr }, { gcn_w[1], 64, 128, nullptr },
    { gcn_w[2],  64, 128, nullptr }, { gcn_w[3], 64, 128, nullptr },
    { fc_w[0], 128, 64, nullptr }, { fc_w[1], 128, 64, nullptr },
    { fc_w[2], 128, 64, nullptr }, { fc_w[3], 128, 64, nullptr },
    { proj_w[0], 320, 64, nullptr }, { proj_w[1], 128, 64, nullptr },
    { proj_w[2], 128, 64, nullptr }, { proj_w[3], 128, 64, nullptr },
  };
  const int NW = sizeof(WS) / sizeof(WS[0]);
  for (int i = 0; i < NW; ++i) {
    WS[i].t = (_Float16*)carve((size_t)WS[i].K * WS[i].Nout * 2);
    int total = WS[i].K * WS[i].Nout;
    k_conv_wt<<<(total + 255) / 256, 256, 0, stream>>>(WS[i].w, WS[i].t, WS[i].K, WS[i].Nout);
  }
  _Float16 *pfc1T = WS[0].t, *pfc2T = WS[1].t, *gfc1T = WS[2].t, *gfc2T = WS[3].t,
           *gprjT = WS[4].t;
  _Float16 *gcnT[4] = { WS[5].t, WS[6].t, WS[7].t, WS[8].t };
  _Float16 *fcT[4]  = { WS[9].t, WS[10].t, WS[11].t, WS[12].t };
  _Float16 *prjT[4] = { WS[13].t, WS[14].t, WS[15].t, WS[16].t };

  float* out_f = (float*)d_out;
  for (int gi = 0; gi < 2; ++gi) {
    const float* x     = (const float*)d_in[0 + 4 * gi];
    const float* f     = (const float*)d_in[1 + 4 * gi];
    const float* gfeat = (const float*)d_in[2 + 4 * gi];
    const int*   edges = (const int*)  d_in[3 + 4 * gi];
    const int*   src   = edges;
    const int*   dst   = edges + E;
    float* feat_out = out_f + (size_t)gi * N * 64;
    float* glob_out = out_f + (size_t)2 * N * 64 + (size_t)gi * G * 64;

    // node branch -> feat[:, 0:128]
    k_node_lin<<<(N * 128 + 255) / 256, 256, 0, stream>>>(x, node_w, node_b, b_feat, 256, N);
    // patch branch: f -> 256 (relu) -> 128 (relu) -> feat[:, 128:256]
    gemm(stream, f, 128, 0, pfc1T, pfc1_b, nullptr, 0, b_cat, 256, 0, 1, N, 256, 128);
    gemm(stream, b_cat, 256, 0, pfc2T, pfc2_b, nullptr, 0, b_feat, 256, 128, 1, N, 128, 256);

    // degrees (with self loops) -> dinv
    k_fill1<<<(N + 255) / 256, 256, 0, stream>>>(dinv, N);
    k_deg_count<<<(E + 255) / 256, 256, 0, stream>>>(dst, dinv, E);
    k_rsqrt_inplace<<<(N + 255) / 256, 256, 0, stream>>>(dinv, N);

    // global branch
    gemm(stream, gfeat, 128, 0, gfc1T, gfc1_b, nullptr, 0, g_t1, 256, 0, 1, G, 256, 128);
    gemm(stream, g_t1, 256, 0, gfc2T, gfc2_b, nullptr, 0, g_t2, 128, 0, 1, G, 128, 256);
    gemm(stream, g_t2, 128, 0, gprjT, gprj_b, nullptr, 0, glob_out, 64, 0, 0, G, 64, 128);

    // 4 PolygonFeatureGathering blocks
    for (int L = 0; L < 4; ++L) {
      const float* h = (L == 0) ? b_feat : b_h;
      const int Kh   = (L == 0) ? 256 : 64;
      // xw = h @ Wg (no bias)
      gemm(stream, h, Kh, 0, gcnT[L], nullptr, nullptr, 0, b_xw, 128, 0, 0, N, 128, Kh);
      // agg = b + self-loop + edge scatter
      k_agg_init<<<(N * 128 + 255) / 256, 256, 0, stream>>>(b_xw, dinv, gcn_b[L], b_agg, N);
      k_gcn_scatter<<<(E * 32 + 255) / 256, 256, 0, stream>>>(src, dst, E, dinv, b_xw, b_agg);
      // out = relu(agg) @ Wfc + bfc
      gemm(stream, b_agg, 128, 1, fcT[L], fc_b[L], nullptr, 0, b_out, 64, 0, 0, N, 64, 128);
      // h' = relu([out | h] @ Wp + bp + out)
      k_concat2<<<(N * (64 + Kh) + 255) / 256, 256, 0, stream>>>(b_out, 64, h, Kh, b_cat, N);
      float* dest = (L == 3) ? feat_out : b_h;
      gemm(stream, b_cat, 64 + Kh, 0, prjT[L], proj_b[L], b_out, 64,
           dest, 64, 0, 1, N, 64, 64 + Kh);
    }
  }
  (void)n_in; (void)out_size; (void)ws_size;
}
